// GroupedVectorCA_38800734552581
// MI455X (gfx1250) — matrix operand order repaired and checked by reading, not compiled, not run on hardware
//
#include <hip/hip_runtime.h>
#include <hip/hip_bf16.h>
#include <math.h>

typedef _Float16 v16h __attribute__((ext_vector_type(16)));
typedef _Float16 h8   __attribute__((ext_vector_type(8)));
typedef float    v8f  __attribute__((ext_vector_type(8)));

#define BN_SCALE 0.99999500003749969f  /* 1/sqrt(1+1e-5) */

// ---------------------------------------------------------------- converts
__global__ void cvt_f16_kernel(const float* __restrict__ x, _Float16* __restrict__ y, int n) {
    int i = blockIdx.x * 256 + threadIdx.x;
    if (i < n) y[i] = (_Float16)x[i];
}

// ---------------------------------------------------------------- fuse constants
// Folds eval-mode BN affines, builds Wfused = Wp2 @ Ww1 (64x8, padded to 64x16 f16),
// bfused = bp2 @ Ww1 + bw1, and h-MLP folded coefficients.
__global__ void fuse_kernel(const float* __restrict__ Wp1, const float* __restrict__ bp1,
                            const float* __restrict__ gp,  const float* __restrict__ btp,
                            const float* __restrict__ Wp2, const float* __restrict__ bp2,
                            const float* __restrict__ Ww1, const float* __restrict__ bw1,
                            const float* __restrict__ bq, const float* __restrict__ gq, const float* __restrict__ btq,
                            const float* __restrict__ bk, const float* __restrict__ gk, const float* __restrict__ btk,
                            const float* __restrict__ bv,
                            _Float16* __restrict__ Wf16, float* __restrict__ bfused,
                            float* __restrict__ qkvA, float* __restrict__ qkvB,
                            float* __restrict__ Whc, float* __restrict__ bhc) {
    int t = threadIdx.x;
    const float s = BN_SCALE;
    if (t < 64) {
        qkvA[0*64+t] = s*gq[t]; qkvB[0*64+t] = bq[t]*s*gq[t] + btq[t];
        qkvA[1*64+t] = s*gk[t]; qkvB[1*64+t] = bk[t]*s*gk[t] + btk[t];
        qkvA[2*64+t] = 1.0f;    qkvB[2*64+t] = bv[t];
        float a = s * gp[t];
        Whc[0*64+t] = Wp1[0*64+t]*a;
        Whc[1*64+t] = Wp1[1*64+t]*a;
        Whc[2*64+t] = Wp1[2*64+t]*a;
        bhc[t] = bp1[t]*a + btp[t];
    }
    if (t < 8) {
        float acc = 0.f;
        for (int j = 0; j < 64; j++) acc += bp2[j] * Ww1[j*8+t];
        bfused[t] = acc + bw1[t];
    }
    for (int idx = t; idx < 64*16; idx += 256) {
        int i = idx >> 4, g = idx & 15;
        float acc = 0.f;
        if (g < 8) for (int j = 0; j < 64; j++) acc += Wp2[i*64+j] * Ww1[j*8+g];
        Wf16[idx] = (_Float16)acc;
    }
}

// ---------------------------------------------------------------- QKV GEMM (WMMA)
// X[R,64] f16 @ W[64,64] f16 -> out[R,64] f32 with folded BN(+ReLU) epilogue.
// grid = (R/16, 3); block = 128 (4 waves, one 16-col tile each).
__global__ void __launch_bounds__(128)
qkv_wmma_kernel(const _Float16* __restrict__ Xh,
                const _Float16* __restrict__ Wq, const _Float16* __restrict__ Wk,
                const _Float16* __restrict__ Wv,
                const float* __restrict__ qkvA, const float* __restrict__ qkvB,
                float* __restrict__ out_q, float* __restrict__ out_k, float* __restrict__ out_v) {
    int sel = blockIdx.y;
    const _Float16* Wh = (sel == 0) ? Wq : ((sel == 1) ? Wk : Wv);
    float* out = (sel == 0) ? out_q : ((sel == 1) ? out_k : out_v);
    int lane = threadIdx.x & 31;
    int wave = threadIdx.x >> 5;            // col tile 0..3
    int r0   = blockIdx.x * 16;
    int col  = wave * 16 + (lane & 15);
    int hi   = lane >> 4;
    v8f acc = {};
    #pragma unroll
    for (int kk = 0; kk < 2; kk++) {
        int kbase = kk * 32;
        v16h a;
        const _Float16* rowp = Xh + (r0 + (lane & 15)) * 64 + kbase + hi * 8;
        h8 lo = *(const h8*)(rowp);
        h8 hv = *(const h8*)(rowp + 16);
        #pragma unroll
        for (int e = 0; e < 8; e++) { a[e] = lo[e]; a[8+e] = hv[e]; }
        v16h b;
        #pragma unroll
        for (int i = 0; i < 8; i++) {
            int k0 = kbase + ((i < 4) ? (hi * 8 + 2 * i) : (16 + hi * 8 + 2 * i - 8));
            b[2*i]   = Wh[(k0  ) * 64 + col];
            b[2*i+1] = Wh[(k0+1) * 64 + col];
        }
        acc = __builtin_amdgcn_wmma_f32_16x16x32_f16(false, a, false, b, (short)0, acc, false, false);
    }
    float alpha = qkvA[sel*64+col], beta = qkvB[sel*64+col];
    bool doRelu = (sel < 2);
    #pragma unroll
    for (int j = 0; j < 8; j++) {
        int r = r0 + hi * 8 + j;
        float v = acc[j] * alpha + beta;
        if (doRelu) v = fmaxf(v, 0.f);
        out[r * 64 + col] = v;
    }
}

// ---------------------------------------------------------------- X @ Ww1 (tiny)
__global__ void proj_ww1_kernel(const float* __restrict__ X, const float* __restrict__ Ww1,
                                float* __restrict__ out, int R) {
    int idx = blockIdx.x * 256 + threadIdx.x;
    if (idx >= R * 8) return;
    int r = idx >> 3, g = idx & 7;
    float acc = 0.f;
    for (int c = 0; c < 64; c++) acc += X[r*64+c] * Ww1[c*8+g];
    out[idx] = acc;
}

// ---------------------------------------------------------------- pass A (WMMA)
// For one row (q-point), iterate 16-col tiles: compute h[16 pairs][64] in-register,
// two v_wmma_f32_16x16x32_f16 against Wfused (64x16), epilogue relu-bn + 8x8 Ww2.
// w[row, col, g] logits written to workspace.
__global__ void __launch_bounds__(256)
passA_kernel(const float* __restrict__ cRow, const float* __restrict__ cCol,
             const float* __restrict__ bQ, const float* __restrict__ aK,
             const _Float16* __restrict__ Wf16, const float* __restrict__ bfused,
             const float* __restrict__ gw, const float* __restrict__ btw,
             const float* __restrict__ Ww2, const float* __restrict__ bw2,
             const float* __restrict__ Whc, const float* __restrict__ bhc,
             int S, float* __restrict__ w) {
    __shared__ float sWh[192], sBh[64], sWw2[64], sBw2[8], sWga[8], sWgb[8];
    __shared__ float tbuf[8][16][8];
    int t = threadIdx.x;
    if (t < 192) sWh[t] = Whc[t];
    if (t < 64)  { sBh[t] = bhc[t]; sWw2[t] = Ww2[t]; }
    if (t < 8)   { sBw2[t] = bw2[t]; sWga[t] = BN_SCALE * gw[t]; sWgb[t] = btw[t]; }
    __syncthreads();

    int lane = t & 31, wave = t >> 5;
    int hi = lane >> 4, p = lane & 15;
    int row = blockIdx.x;
    float rx = cRow[row*3+0], ry = cRow[row*3+1], rz = cRow[row*3+2];

    // Preload B fragments (Wfused 64x16) for both K-steps.
    v16h b0, b1;
    int colB = lane & 15;
    #pragma unroll
    for (int i = 0; i < 8; i++) {
        int k0 = (i < 4) ? (hi * 8 + 2 * i) : (16 + hi * 8 + 2 * i - 8);
        b0[2*i]   = Wf16[(k0     ) * 16 + colB];
        b0[2*i+1] = Wf16[(k0 +  1) * 16 + colB];
        b1[2*i]   = Wf16[(k0 + 32) * 16 + colB];
        b1[2*i+1] = Wf16[(k0 + 33) * 16 + colB];
    }
    int g = lane & 15;
    float bq_g  = (g < 8) ? bQ[row*8+g] : 0.f;
    float bf_g  = (g < 8) ? bfused[g]   : 0.f;
    float wga_g = (g < 8) ? sWga[g]     : 0.f;
    float wgb_g = (g < 8) ? sWgb[g]     : 0.f;

    int ntiles = S >> 4;                    // 32 or 64, divisible by 8 waves
    for (int tile = wave; tile < ntiles; tile += 8) {
        int m0 = tile * 16;
        int m  = m0 + p;
        float dx = rx - cCol[m*3+0], dy = ry - cCol[m*3+1], dz = rz - cCol[m*3+2];
        v16h a0, a1;
        #pragma unroll
        for (int e = 0; e < 16; e++) {
            int c0 = (e < 8) ? (hi*8 + e) : (16 + hi*8 + (e - 8));
            float h0 = fmaxf(dx*sWh[c0] + dy*sWh[64+c0] + dz*sWh[128+c0] + sBh[c0], 0.f);
            a0[e] = (_Float16)h0;
            int c1 = c0 + 32;
            float h1 = fmaxf(dx*sWh[c1] + dy*sWh[64+c1] + dz*sWh[128+c1] + sBh[c1], 0.f);
            a1[e] = (_Float16)h1;
        }
        v8f acc = {};
        acc = __builtin_amdgcn_wmma_f32_16x16x32_f16(false, a0, false, b0, (short)0, acc, false, false);
        acc = __builtin_amdgcn_wmma_f32_16x16x32_f16(false, a1, false, b1, (short)0, acc, false, false);
        if (g < 8) {
            #pragma unroll
            for (int j = 0; j < 8; j++) {
                int pr = hi * 8 + j;
                float pre = acc[j] + aK[(m0 + pr)*8 + g] - bq_g + bf_g;
                tbuf[wave][pr][g] = fmaxf(pre * wga_g + wgb_g, 0.f);
            }
        }
        __syncthreads();
        #pragma unroll
        for (int q4 = 0; q4 < 4; q4++) {
            int idx = lane * 4 + q4;
            int pr = idx >> 3, go = idx & 7;
            float acc2 = sBw2[go];
            #pragma unroll
            for (int gg = 0; gg < 8; gg++) acc2 += tbuf[wave][pr][gg] * sWw2[gg*8+go];
            w[((long)row * S + m0 + pr) * 8 + go] = acc2;
        }
        __syncthreads();
    }
}

// ---------------------------------------------------------------- pass B1
// Row softmax over S cols + grouped contraction:
// o[row, g*8+i] = sum_m wm[m,g]*vCol[m,g*8+i] + (sum_m wm[m,g]*h[m,:]) @ Wp2[:, g*8+i] + bp2
__global__ void __launch_bounds__(256)
passB1_kernel(const float* __restrict__ w, const float* __restrict__ cRow,
              const float* __restrict__ cCol, const float* __restrict__ vCol,
              const float* __restrict__ Whc, const float* __restrict__ bhc,
              const float* __restrict__ Wp2, const float* __restrict__ bp2,
              int S, float* __restrict__ o) {
    __shared__ float wm[8192];     // up to 1024 x 8
    __shared__ float cc[3072];     // up to 1024 x 3
    __shared__ float red[2048];    // 256 x 8 reduction
    __shared__ float sAcc[512];    // 8 x 64
    __shared__ float hcoef[256];   // w0,w1,w2,b  (4 x 64)
    int t = threadIdx.x, row = blockIdx.x;
    for (int i = t; i < S * 3; i += 256) cc[i] = cCol[i];
    if (t < 64) { hcoef[t] = Whc[t]; hcoef[64+t] = Whc[64+t]; hcoef[128+t] = Whc[128+t]; hcoef[192+t] = bhc[t]; }
    __syncthreads();
    float rx = cRow[row*3+0], ry = cRow[row*3+1], rz = cRow[row*3+2];
    const float* wrow = w + (long)row * S * 8;

    float lmax[8];
    #pragma unroll
    for (int g = 0; g < 8; g++) lmax[g] = -3.4e38f;
    for (int m = t; m < S; m += 256)
        #pragma unroll
        for (int g = 0; g < 8; g++) lmax[g] = fmaxf(lmax[g], wrow[m*8+g]);
    #pragma unroll
    for (int g = 0; g < 8; g++) red[t*8+g] = lmax[g];
    __syncthreads();
    for (int off = 128; off > 0; off >>= 1) {
        if (t < off)
            #pragma unroll
            for (int g = 0; g < 8; g++) red[t*8+g] = fmaxf(red[t*8+g], red[(t+off)*8+g]);
        __syncthreads();
    }
    float gmax[8];
    #pragma unroll
    for (int g = 0; g < 8; g++) gmax[g] = red[g];
    __syncthreads();

    float lsum[8] = {0,0,0,0,0,0,0,0};
    for (int m = t; m < S; m += 256)
        #pragma unroll
        for (int g = 0; g < 8; g++) {
            float e = __expf(wrow[m*8+g] - gmax[g]);
            wm[m*8+g] = e; lsum[g] += e;
        }
    #pragma unroll
    for (int g = 0; g < 8; g++) red[t*8+g] = lsum[g];
    __syncthreads();
    for (int off = 128; off > 0; off >>= 1) {
        if (t < off)
            #pragma unroll
            for (int g = 0; g < 8; g++) red[t*8+g] += red[(t+off)*8+g];
        __syncthreads();
    }
    float ginv[8];
    #pragma unroll
    for (int g = 0; g < 8; g++) ginv[g] = 1.f / red[g];
    __syncthreads();
    for (int m = t; m < S; m += 256)
        #pragma unroll
        for (int g = 0; g < 8; g++) wm[m*8+g] *= ginv[g];
    __syncthreads();

    // s[g][c] = sum_m wm[m,g] * h(m)[c]   (each thread: one channel, two groups)
    {
        int c = t & 63, gb = t >> 6;   // gb 0..3 -> groups gb, gb+4
        float w0 = hcoef[c], w1 = hcoef[64+c], w2c = hcoef[128+c], b0 = hcoef[192+c];
        float a0 = 0.f, a1 = 0.f;
        for (int m = 0; m < S; m++) {
            float dx = rx - cc[m*3+0], dy = ry - cc[m*3+1], dz = rz - cc[m*3+2];
            float h = fmaxf(dx*w0 + dy*w1 + dz*w2c + b0, 0.f);
            a0 += wm[m*8+gb] * h;
            a1 += wm[m*8+gb+4] * h;
        }
        sAcc[gb*64 + c]     = a0;
        sAcc[(gb+4)*64 + c] = a1;
    }
    __syncthreads();

    if (t < 64) {
        int g = t >> 3, ch = t;   // ch = g*8 + i
        float acc = bp2[ch];      // sum_m wm == 1 -> bias passes through
        for (int m = 0; m < S; m++) acc += wm[m*8+g] * vCol[m*64+ch];
        for (int c = 0; c < 64; c++) acc += sAcc[g*64+c] * Wp2[c*64+ch];
        o[row*64 + ch] = acc;
    }
}

// ---------------------------------------------------------------- pass B2
// Column softmax over R rows, mean over G -> wo[r, col]
__global__ void __launch_bounds__(256)
passB2_kernel(const float* __restrict__ w, int R, int S, float* __restrict__ wo) {
    __shared__ float red[2048];
    int t = threadIdx.x, col = blockIdx.x;
    float lmax[8];
    #pragma unroll
    for (int g = 0; g < 8; g++) lmax[g] = -3.4e38f;
    for (int r = t; r < R; r += 256)
        #pragma unroll
        for (int g = 0; g < 8; g++) lmax[g] = fmaxf(lmax[g], w[((long)r*S+col)*8+g]);
    #pragma unroll
    for (int g = 0; g < 8; g++) red[t*8+g] = lmax[g];
    __syncthreads();
    for (int off = 128; off > 0; off >>= 1) {
        if (t < off)
            #pragma unroll
            for (int g = 0; g < 8; g++) red[t*8+g] = fmaxf(red[t*8+g], red[(t+off)*8+g]);
        __syncthreads();
    }
    float gmax[8];
    #pragma unroll
    for (int g = 0; g < 8; g++) gmax[g] = red[g];
    __syncthreads();
    float lsum[8] = {0,0,0,0,0,0,0,0};
    for (int r = t; r < R; r += 256)
        #pragma unroll
        for (int g = 0; g < 8; g++) lsum[g] += __expf(w[((long)r*S+col)*8+g] - gmax[g]);
    #pragma unroll
    for (int g = 0; g < 8; g++) red[t*8+g] = lsum[g];
    __syncthreads();
    for (int off = 128; off > 0; off >>= 1) {
        if (t < off)
            #pragma unroll
            for (int g = 0; g < 8; g++) red[t*8+g] += red[(t+off)*8+g];
        __syncthreads();
    }
    float ginv[8];
    #pragma unroll
    for (int g = 0; g < 8; g++) ginv[g] = 1.f / red[g];
    for (int r = t; r < R; r += 256) {
        float acc = 0.f;
        #pragma unroll
        for (int g = 0; g < 8; g++) acc += __expf(w[((long)r*S+col)*8+g] - gmax[g]) * ginv[g];
        wo[(long)r*S + col] = acc * 0.125f;
    }
}

// ---------------------------------------------------------------- host
extern "C" void kernel_launch(void* const* d_in, const int* in_sizes, int n_in,
                              void* d_out, int out_size, void* d_ws, size_t ws_size,
                              hipStream_t stream) {
    (void)in_sizes; (void)n_in; (void)out_size; (void)ws_size;
    const int N = 1024, M = 512, C = 64;

    const float* f1  = (const float*)d_in[0];
    const float* c1  = (const float*)d_in[1];
    const float* f2  = (const float*)d_in[2];
    const float* c2  = (const float*)d_in[3];
    const float* Wq  = (const float*)d_in[4];  const float* bq  = (const float*)d_in[5];
    const float* gq  = (const float*)d_in[6];  const float* btq = (const float*)d_in[7];
    const float* Wk  = (const float*)d_in[8];  const float* bk  = (const float*)d_in[9];
    const float* gk  = (const float*)d_in[10]; const float* btk = (const float*)d_in[11];
    const float* Wv  = (const float*)d_in[12]; const float* bv  = (const float*)d_in[13];
    const float* Wp1 = (const float*)d_in[14]; const float* bp1 = (const float*)d_in[15];
    const float* gp  = (const float*)d_in[16]; const float* btp = (const float*)d_in[17];
    const float* Wp2 = (const float*)d_in[18]; const float* bp2 = (const float*)d_in[19];
    const float* Ww1 = (const float*)d_in[20]; const float* bw1 = (const float*)d_in[21];
    const float* gw  = (const float*)d_in[22]; const float* btw = (const float*)d_in[23];
    const float* Ww2 = (const float*)d_in[24]; const float* bw2 = (const float*)d_in[25];

    char* base = (char*)d_ws; size_t off = 0;
    auto alloc = [&](size_t bytes) -> char* {
        char* p = base + off; off = (off + bytes + 255) & ~(size_t)255; return p;
    };
    float*    q1   = (float*)alloc((size_t)N*C*4);
    float*    k1   = (float*)alloc((size_t)N*C*4);
    float*    v1   = (float*)alloc((size_t)N*C*4);
    float*    q2   = (float*)alloc((size_t)M*C*4);
    float*    k2   = (float*)alloc((size_t)M*C*4);
    float*    v2   = (float*)alloc((size_t)M*C*4);
    _Float16* f1h  = (_Float16*)alloc((size_t)N*C*2);
    _Float16* f2h  = (_Float16*)alloc((size_t)M*C*2);
    _Float16* Wqh  = (_Float16*)alloc((size_t)C*C*2);
    _Float16* Wkh  = (_Float16*)alloc((size_t)C*C*2);
    _Float16* Wvh  = (_Float16*)alloc((size_t)C*C*2);
    _Float16* Wf16 = (_Float16*)alloc(64*16*2);
    float*    bfu  = (float*)alloc(8*4);
    float*    qA   = (float*)alloc(3*64*4);
    float*    qB   = (float*)alloc(3*64*4);
    float*    Whc  = (float*)alloc(3*64*4);
    float*    bhc  = (float*)alloc(64*4);
    float*    bQ1  = (float*)alloc((size_t)N*8*4);
    float*    aK1  = (float*)alloc((size_t)N*8*4);
    float*    bQ2  = (float*)alloc((size_t)M*8*4);
    float*    aK2  = (float*)alloc((size_t)M*8*4);
    float*    w1   = (float*)alloc((size_t)N*M*8*4);
    float*    w2   = (float*)alloc((size_t)M*N*8*4);

    float* o1  = (float*)d_out;
    float* wo1 = o1  + (size_t)N*C;
    float* o2  = wo1 + (size_t)N*M;
    float* wo2 = o2  + (size_t)M*C;

    cvt_f16_kernel<<<(N*C+255)/256, 256, 0, stream>>>(f1, f1h, N*C);
    cvt_f16_kernel<<<(M*C+255)/256, 256, 0, stream>>>(f2, f2h, M*C);
    cvt_f16_kernel<<<(C*C+255)/256, 256, 0, stream>>>(Wq, Wqh, C*C);
    cvt_f16_kernel<<<(C*C+255)/256, 256, 0, stream>>>(Wk, Wkh, C*C);
    cvt_f16_kernel<<<(C*C+255)/256, 256, 0, stream>>>(Wv, Wvh, C*C);

    fuse_kernel<<<1, 256, 0, stream>>>(Wp1, bp1, gp, btp, Wp2, bp2, Ww1, bw1,
                                       bq, gq, btq, bk, gk, btk, bv,
                                       Wf16, bfu, qA, qB, Whc, bhc);

    qkv_wmma_kernel<<<dim3(N/16, 3), 128, 0, stream>>>(f1h, Wqh, Wkh, Wvh, qA, qB, q1, k1, v1);
    qkv_wmma_kernel<<<dim3(M/16, 3), 128, 0, stream>>>(f2h, Wqh, Wkh, Wvh, qA, qB, q2, k2, v2);

    proj_ww1_kernel<<<(N*8+255)/256, 256, 0, stream>>>(q1, Ww1, bQ1, N);
    proj_ww1_kernel<<<(N*8+255)/256, 256, 0, stream>>>(k1, Ww1, aK1, N);
    proj_ww1_kernel<<<(M*8+255)/256, 256, 0, stream>>>(q2, Ww1, bQ2, M);
    proj_ww1_kernel<<<(M*8+255)/256, 256, 0, stream>>>(k2, Ww1, aK2, M);

    passA_kernel<<<N, 256, 0, stream>>>(c1, c2, bQ1, aK2, Wf16, bfu, gw, btw, Ww2, bw2, Whc, bhc, M, w1);
    passA_kernel<<<M, 256, 0, stream>>>(c2, c1, bQ2, aK1, Wf16, bfu, gw, btw, Ww2, bw2, Whc, bhc, N, w2);

    passB1_kernel<<<N, 256, 0, stream>>>(w1, c1, c2, v2, Whc, bhc, Wp2, bp2, M, o1);
    passB1_kernel<<<M, 256, 0, stream>>>(w2, c2, c1, v1, Whc, bhc, Wp2, bp2, N, o2);

    passB2_kernel<<<M, 256, 0, stream>>>(w1, N, M, wo1);
    passB2_kernel<<<N, 256, 0, stream>>>(w2, M, N, wo2);
}
